// HeteroGNN_75247827026323
// MI455X (gfx1250) — matrix operand order repaired and checked by reading, not compiled, not run on hardware
//
#include <hip/hip_runtime.h>

typedef __attribute__((ext_vector_type(16))) _Float16 v16h;
typedef __attribute__((ext_vector_type(8)))  float    v8f;

#define NUSER 100000
#define NITEM 30000

// ---------------------------------------------------------------------------
// Count in-degree per destination node (float, for mean division).
// ---------------------------------------------------------------------------
__global__ void k_count(const int* __restrict__ dst, int E, float* __restrict__ cnt) {
    int i = blockIdx.x * blockDim.x + threadIdx.x;
    if (i < E) atomicAdd(&cnt[dst[i]], 1.0f);
}

// ---------------------------------------------------------------------------
// Edge scatter: agg[dst] += x[src]. One block = EPB edges; edge indices are
// staged in LDS once per block (kills the redundant per-thread index loads),
// then each thread moves one float4 of the feature row via f32 global atomics.
// ---------------------------------------------------------------------------
template <int D>
__global__ void k_scatter(const float* __restrict__ x, const int* __restrict__ src,
                          const int* __restrict__ dst, int E, float* __restrict__ agg) {
    constexpr int CH  = D / 4;     // float4 chunks per edge
    constexpr int EPB = 256 / CH;  // edges per block
    __shared__ int ss[EPB], sd[EPB];

    const int e0 = blockIdx.x * EPB;
    const int t  = threadIdx.x;
    if (t < EPB) {
        int e = e0 + t;
        ss[t] = (e < E) ? src[e] : 0;
    } else if (t < 2 * EPB) {
        int i = t - EPB;
        int e = e0 + i;
        sd[i] = (e < E) ? dst[e] : 0;
    }
    __syncthreads();

    const int el = t / CH;
    const int c  = t % CH;
    const int e  = e0 + el;
    if (e >= E) return;

    const int s  = ss[el];
    const int dd = sd[el];
    const float4 v = ((const float4*)(x + (size_t)s * D))[c];
    float* o = agg + (size_t)dd * D + 4 * c;
    atomicAdd(o + 0, v.x);
    atomicAdd(o + 1, v.y);
    atomicAdd(o + 2, v.z);
    atomicAdd(o + 3, v.w);
}

// ---------------------------------------------------------------------------
// Normalize: agg[row, :] /= max(cnt[row], 1)
// ---------------------------------------------------------------------------
__global__ void k_mean(float* __restrict__ agg, const float* __restrict__ cnt,
                       int n, int d) {
    long long t = (long long)blockIdx.x * blockDim.x + threadIdx.x;
    if (t >= (long long)n * d) return;
    int row = (int)(t / d);
    float c = cnt[row];
    c = c > 1.0f ? c : 1.0f;
    agg[t] = agg[t] / c;
}

// ---------------------------------------------------------------------------
// Convert 8 consecutive f32 to f16 (two float4 loads).
// ---------------------------------------------------------------------------
__device__ __forceinline__ void cvt8(const float* __restrict__ p, _Float16* __restrict__ o) {
    float4 a = ((const float4*)p)[0];
    float4 b = ((const float4*)p)[1];
    o[0] = (_Float16)a.x; o[1] = (_Float16)a.y;
    o[2] = (_Float16)a.z; o[3] = (_Float16)a.w;
    o[4] = (_Float16)b.x; o[5] = (_Float16)b.y;
    o[6] = (_Float16)b.z; o[7] = (_Float16)b.w;
}

// ---------------------------------------------------------------------------
// Fused SAGE linear: out[n,:] = mean[n,:] @ Wl + bias + xdst[n,:] @ Wr
// as [mean | xdst] (M x 2DIN) @ [Wl ; Wr] (2DIN x DOUT) using
// v_wmma_f32_16x16x32_f16. Block = 128 threads = 4 waves stacked over M
// (64 rows x 16 cols per block). The 16-column weight panel (both Wl and Wr
// segments) is staged once per block in LDS as f16; the K loop is fully
// unrolled (KTOT/32 back-to-back WMMAs). mode: 0=write, 1=write+relu,
// 2=out=0.5*(out+res), 3=relu(0.5*(out+res)).
// ---------------------------------------------------------------------------
template <int DIN, int DOUT>
__global__ void k_sage_wmma(const float* __restrict__ mean, const float* __restrict__ xd,
                            const float* __restrict__ Wl, const float* __restrict__ Wr,
                            const float* __restrict__ bias, float* __restrict__ out,
                            int M, int mode) {
    constexpr int KTOT = 2 * DIN;
    __shared__ _Float16 sW[KTOT * 16];   // [k][n], n = 0..15 within panel

    const int colBase = blockIdx.y * 16;

    // ---- Cooperative, coalesced weight panel stage: global f32 -> LDS f16 ----
    for (int i = threadIdx.x; i < KTOT * 16; i += 128) {
        int k = i >> 4;
        int n = i & 15;
        float v = (k < DIN) ? Wl[(size_t)k * DOUT + colBase + n]
                            : Wr[(size_t)(k - DIN) * DOUT + colBase + n];
        sW[i] = (_Float16)v;
    }
    __syncthreads();

    const int lane    = threadIdx.x & 31;
    const int wave    = threadIdx.x >> 5;
    const int rowBase = blockIdx.x * 64 + wave * 16;
    const int r15     = lane & 15;
    const int row     = rowBase + r15;        // A row held by this lane
    const bool rowOk  = row < M;
    const int kOff    = (lane < 16) ? 0 : 8;  // A K-group offset per ISA layout

    v8f acc = {};

#pragma unroll
    for (int kt = 0; kt < KTOT; kt += 32) {
        // ---- A fragment: lane holds row r15, K = kt+kOff+{0..7, 16..23}.
        //      kt is compile-time after unroll, so the segment select
        //      (mean vs xdst) is resolved statically per iteration.
        v16h a;
        if (rowOk) {
            const float* abase = (kt < DIN)
                ? (mean + (size_t)row * DIN + (kt + kOff))
                : (xd   + (size_t)row * DIN + (kt - DIN + kOff));
            _Float16 ta[16];
            cvt8(abase,      ta);
            cvt8(abase + 16, ta + 8);
#pragma unroll
            for (int j = 0; j < 16; ++j) a[j] = ta[j];
        } else {
#pragma unroll
            for (int j = 0; j < 16; ++j) a[j] = (_Float16)0.0f;
        }

        // ---- B fragment from LDS: lane = N col r15, K = kt+kOff+{0..7, 16..23} ----
        const _Float16* bp = &sW[(kt + kOff) * 16 + r15];
        v16h b;
#pragma unroll
        for (int j = 0; j < 16; ++j) b[j] = bp[((j < 8) ? j : j + 8) * 16];

        acc = __builtin_amdgcn_wmma_f32_16x16x32_f16(
            /*neg_a=*/false, a, /*neg_b=*/false, b,
            /*c_mod=*/(short)0, acc, /*reuse_a=*/false, /*reuse_b=*/false);
    }

    // ---- Epilogue: C/D layout (lane = N col, reg r = M row, +8 upper) ----
    const int col  = colBase + r15;
    const float bv = bias[col];
    const int rAdd = (lane < 16) ? 0 : 8;
#pragma unroll
    for (int r = 0; r < 8; ++r) {
        int orow = rowBase + rAdd + r;
        if (orow >= M) continue;
        float v = acc[r] + bv;
        size_t oi = (size_t)orow * DOUT + col;
        if (mode == 0)      out[oi] = v;
        else if (mode == 1) out[oi] = fmaxf(v, 0.0f);
        else if (mode == 2) out[oi] = 0.5f * (out[oi] + v);
        else                out[oi] = fmaxf(0.5f * (out[oi] + v), 0.0f);
    }
}

// ---------------------------------------------------------------------------
// One hetero layer (templated on dims so device code fully specializes).
// ---------------------------------------------------------------------------
template <int DIN, int DOUT>
static void run_layer(const float* xu, const float* xi,
                      const int* er, int ER_, const int* eb, int ERB_,
                      const int* es, int ES_,
                      const float* Wl, const float* bl, const float* Wr,
                      float* aggU, float* aggIr, float* aggIs,
                      const float* cntU, const float* cntIr, const float* cntIs,
                      float* ou, float* oi, bool relu, hipStream_t stream) {
    const int TPB = 256;

    // Zero aggregation buffers
    hipMemsetAsync(aggU,  0, (size_t)NUSER * DIN * sizeof(float), stream);
    hipMemsetAsync(aggIr, 0, (size_t)NITEM * DIN * sizeof(float), stream);
    hipMemsetAsync(aggIs, 0, (size_t)NITEM * DIN * sizeof(float), stream);

    // Scatter (segment-sum via f32 global atomics, LDS-staged edge indices)
    constexpr int EPB = 256 / (DIN / 4);
    k_scatter<DIN><<<(ER_  + EPB - 1) / EPB, TPB, 0, stream>>>(xu, er, er + ER_,  ER_,  aggIr);
    k_scatter<DIN><<<(ERB_ + EPB - 1) / EPB, TPB, 0, stream>>>(xi, eb, eb + ERB_, ERB_, aggU);
    k_scatter<DIN><<<(ES_  + EPB - 1) / EPB, TPB, 0, stream>>>(xi, es, es + ES_,  ES_,  aggIs);

    // Mean
    long long nu = (long long)NUSER * DIN, ni = (long long)NITEM * DIN;
    k_mean<<<(unsigned)((nu + TPB - 1) / TPB), TPB, 0, stream>>>(aggU,  cntU,  NUSER, DIN);
    k_mean<<<(unsigned)((ni + TPB - 1) / TPB), TPB, 0, stream>>>(aggIr, cntIr, NITEM, DIN);
    k_mean<<<(unsigned)((ni + TPB - 1) / TPB), TPB, 0, stream>>>(aggIs, cntIs, NITEM, DIN);

    // WMMA GEMMs. Relation order in params: 0=rates, 1=rated_by, 2=similar.
    const size_t wstride = (size_t)DIN * DOUT;
    dim3 blk(128);
    dim3 gU((NUSER + 63) / 64, DOUT / 16);
    dim3 gI((NITEM + 63) / 64, DOUT / 16);
    // user output: single relation (rated_by)
    k_sage_wmma<DIN, DOUT><<<gU, blk, 0, stream>>>(
        aggU, xu, Wl + 1 * wstride, Wr + 1 * wstride, bl + 1 * DOUT, ou, NUSER,
        relu ? 1 : 0);
    // item output: rates (write), then similar (combine 0.5*(a+b) [+relu])
    k_sage_wmma<DIN, DOUT><<<gI, blk, 0, stream>>>(
        aggIr, xi, Wl + 0 * wstride, Wr + 0 * wstride, bl + 0 * DOUT, oi, NITEM, 0);
    k_sage_wmma<DIN, DOUT><<<gI, blk, 0, stream>>>(
        aggIs, xi, Wl + 2 * wstride, Wr + 2 * wstride, bl + 2 * DOUT, oi, NITEM,
        relu ? 3 : 2);
}

// ---------------------------------------------------------------------------
// Host launch
// ---------------------------------------------------------------------------
extern "C" void kernel_launch(void* const* d_in, const int* in_sizes, int n_in,
                              void* d_out, int out_size, void* d_ws, size_t ws_size,
                              hipStream_t stream) {
    const float* x_user = (const float*)d_in[0];
    const float* x_item = (const float*)d_in[1];
    const int* er = (const int*)d_in[2]; const int ER_  = in_sizes[2] / 2;
    const int* eb = (const int*)d_in[3]; const int ERB_ = in_sizes[3] / 2;
    const int* es = (const int*)d_in[4]; const int ES_  = in_sizes[4] / 2;
    const float* Wl0 = (const float*)d_in[5];
    const float* bl0 = (const float*)d_in[6];
    const float* Wr0 = (const float*)d_in[7];
    const float* Wl1 = (const float*)d_in[8];
    const float* bl1 = (const float*)d_in[9];
    const float* Wr1 = (const float*)d_in[10];
    const float* Wl2 = (const float*)d_in[11];
    const float* bl2 = (const float*)d_in[12];
    const float* Wr2 = (const float*)d_in[13];
    float* out = (float*)d_out;

    // Workspace layout (floats)
    float* ws = (float*)d_ws;
    size_t off = 0;
    float* cntU  = ws + off; off += NUSER;
    float* cntIr = ws + off; off += NITEM;
    float* cntIs = ws + off; off += NITEM;
    float* aggU  = ws + off; off += (size_t)NUSER * 128;
    float* aggIr = ws + off; off += (size_t)NITEM * 128;
    float* aggIs = ws + off; off += (size_t)NITEM * 128;
    float* hu0   = ws + off; off += (size_t)NUSER * 128;
    float* hi0   = ws + off; off += (size_t)NITEM * 128;
    float* hu1   = ws + off; off += (size_t)NUSER * 128;
    float* hi1   = ws + off; off += (size_t)NITEM * 128;

    const int TPB = 256;

    // Degrees depend only on dst indices -> compute once per call.
    hipMemsetAsync(cntU, 0, (size_t)(NUSER + 2 * NITEM) * sizeof(float), stream);
    k_count<<<(ERB_ + TPB - 1) / TPB, TPB, 0, stream>>>(eb + ERB_, ERB_, cntU);
    k_count<<<(ER_  + TPB - 1) / TPB, TPB, 0, stream>>>(er + ER_,  ER_,  cntIr);
    k_count<<<(ES_  + TPB - 1) / TPB, TPB, 0, stream>>>(es + ES_,  ES_,  cntIs);

    // Layer 0: F=64 -> H=128, ReLU
    run_layer<64, 128>(x_user, x_item, er, ER_, eb, ERB_, es, ES_,
                       Wl0, bl0, Wr0, aggU, aggIr, aggIs, cntU, cntIr, cntIs,
                       hu0, hi0, true, stream);
    // Layer 1: H=128 -> H=128, ReLU
    run_layer<128, 128>(hu0, hi0, er, ER_, eb, ERB_, es, ES_,
                        Wl1, bl1, Wr1, aggU, aggIr, aggIs, cntU, cntIr, cntIs,
                        hu1, hi1, true, stream);
    // Layer 2: H=128 -> O=64, no ReLU, write straight into d_out = [u ; i]
    run_layer<128, 64>(hu1, hi1, er, ER_, eb, ERB_, es, ES_,
                       Wl2, bl2, Wr2, aggU, aggIr, aggIs, cntU, cntIr, cntIs,
                       out, out + (size_t)NUSER * 64, false, stream);
}
